// GRUAutoencoder_81930796138844
// MI455X (gfx1250) — matrix operand order repaired and checked by reading, not compiled, not run on hardware
//
#include <hip/hip_runtime.h>
#include <hip/hip_bf16.h>
#include <math.h>

typedef __attribute__((ext_vector_type(16))) _Float16 v16h;
typedef __attribute__((ext_vector_type(8)))  float    v8f;

#define B_SZ   512
#define T_SZ   512
#define I_DIMC 128
#define H_DIMC 64
#define G3     192
#define BTILE  16

// ---- dynamic LDS byte offsets (WMMA fragment regions 32B-aligned) ----
#define L_WB0   0u        // gate-input weight B-frags: enc Wih 12*4*512 halves (49152B); dec Wcomb 12*2*512
#define L_WB1   49152u    // hidden weight B-frags: 12*2*512 halves (24576B)
#define L_FCX   73728u    // enc: x A-frags 4*512 halves (4096B); dec: fc_W B-frags 8*2*512 halves (16384B)
#define L_HFR   90112u    // h A-frags: 2*512 halves (2048B)
#define L_BI    92160u    // f32[192] gi bias (enc_bih / bcomb)
#define L_BH    92928u    // f32[192] gh bias
#define L_BI0   93696u    // f32[192] dec_bih (decoder t==0)
#define L_FCB   94464u    // f32[128] fc_b
#define L_TOTAL 94976u

#define WMMA(A, Bv, C) \
  __builtin_amdgcn_wmma_f32_16x16x32_f16(false, (A), false, (Bv), (short)0, (C), false, false)

__device__ __forceinline__ float fast_sigmoid(float v) {
  return __builtin_amdgcn_rcpf(1.0f + __expf(-v));
}
__device__ __forceinline__ float fast_tanh(float y) {
  y = fminf(fmaxf(y, -15.0f), 15.0f);
  float e = __expf(2.0f * y);
  return 1.0f - 2.0f * __builtin_amdgcn_rcpf(e + 1.0f);
}

// fold Wcomb = dec_Wih(192x128) @ fc_W(128x64);  bcomb = dec_bih + dec_Wih @ fc_b
__global__ void gruae_prep(const float* __restrict__ dWih, const float* __restrict__ fcW,
                           const float* __restrict__ fcb,  const float* __restrict__ dbih,
                           float* __restrict__ wcomb, float* __restrict__ bcomb) {
  int n = blockIdx.x;    // 0..191
  int h = threadIdx.x;   // 0..63
  float acc = 0.0f;
  for (int j = 0; j < I_DIMC; ++j) acc += dWih[n * I_DIMC + j] * fcW[j * H_DIMC + h];
  wcomb[n * H_DIMC + h] = acc;
  if (h == 0) {
    float b = dbih[n];
    for (int j = 0; j < I_DIMC; ++j) b += dWih[n * I_DIMC + j] * fcb[j];
    bcomb[n] = b;
  }
}

__global__ __launch_bounds__(128) void gruae_main(
    const float* __restrict__ x,
    const float* __restrict__ eWih, const float* __restrict__ eWhh,
    const float* __restrict__ ebih, const float* __restrict__ ebhh,
    const float* __restrict__ dWhh,
    const float* __restrict__ dbih, const float* __restrict__ dbhh,
    const float* __restrict__ fcW,  const float* __restrict__ fcb,
    const float* __restrict__ wcomb, const float* __restrict__ bcomb,
    float* __restrict__ out) {
  extern __shared__ char smem[];
  _Float16* wb0 = (_Float16*)(smem + L_WB0);
  _Float16* wb1 = (_Float16*)(smem + L_WB1);
  _Float16* fcx = (_Float16*)(smem + L_FCX);
  _Float16* hfr = (_Float16*)(smem + L_HFR);
  float*    bI  = (float*)(smem + L_BI);
  float*    bH  = (float*)(smem + L_BH);
  float*    bI0 = (float*)(smem + L_BI0);
  float*    fb  = (float*)(smem + L_FCB);

  const int tid  = threadIdx.x;
  const int lane = tid & 31;
  const int wv   = tid >> 5;             // wave 0..3: owns gate tiles {wv, wv+4, wv+8}
  const int b0   = blockIdx.x * BTILE;   // batch-row tile base
  const int nsub = lane & 15;            // C/D column within 16-wide tile
  const int mhi  = (lane >> 4) << 3;     // C/D row offset (VGPR i -> row mhi+i)
  const int hcol = (wv << 4) | nsub;     // this lane's gate/h column (0..63)

  // ---- stage encoder weights as WMMA B-fragments (f32 -> f16) ----
  for (int idx = tid; idx < G3 * I_DIMC; idx += 128) {          // Wih[192][128], 4 K-chunks
    int n = idx >> 7, k = idx & 127;
    int l = (((k >> 4) & 1) << 4) | (n & 15);
    wb0[((((n >> 4) << 2) | (k >> 5)) * 32 + l) * 16 + (k & 15)] = (_Float16)eWih[idx];
  }
  for (int idx = tid; idx < G3 * H_DIMC; idx += 128) {          // Whh[192][64], 2 K-chunks
    int n = idx >> 6, k = idx & 63;
    int l = (((k >> 4) & 1) << 4) | (n & 15);
    wb1[((((n >> 4) << 1) | (k >> 5)) * 32 + l) * 16 + (k & 15)] = (_Float16)eWhh[idx];
  }
  for (int idx = tid; idx < G3; idx += 128) { bI[idx] = ebih[idx]; bH[idx] = ebhh[idx]; }
  for (int idx = tid; idx < 2 * 512; idx += 128) hfr[idx] = (_Float16)0.0f;   // h0 = 0
  __syncthreads();

  // h state: this lane's (rows mhi..mhi+7, col hcol) slice lives in registers forever
  float hprev[8];
  #pragma unroll
  for (int i = 0; i < 8; ++i) hprev[i] = 0.0f;

  // hoisted h-frag store addressing for this lane's column
  _Float16* hdst;
  {
    int kk = hcol & 31, c = hcol >> 5;
    int ii = kk & 15;
    int lb = (kk >> 4) << 4;
    hdst = hfr + (c * 32 + lb + mhi) * 16 + ii;   // + i*16 per row
  }
  // hoisted x staging addressing: thread tid handles column k = tid for rows m=0..15
  const float* xsrc0 = x + (size_t)b0 * T_SZ * I_DIMC + tid;
  _Float16* xdst;
  {
    int kk = tid & 31;
    xdst = fcx + ((tid >> 5) * 32 + ((kk >> 4) << 4)) * 16 +
           (kk & 15);                             // + m*16 per row
  }

  // encoder gate biases hoisted into registers (loop-invariant; avoids per-step
  // LDS reloads forced by may-alias analysis against the hfr stores)
  const float eb_r  = bI[hcol]       + bH[hcol];
  const float eb_z  = bI[64 + hcol]  + bH[64 + hcol];
  const float eb_ni = bI[128 + hcol];
  const float eb_nh = bH[128 + hcol];

  // ================= encoder: 512 sequential steps =================
  for (int t = 0; t < T_SZ; ++t) {
    const float* xs = xsrc0 + (size_t)t * I_DIMC;
    #pragma unroll
    for (int m = 0; m < 16; ++m)
      xdst[m * 16] = (_Float16)__builtin_nontemporal_load(xs + (size_t)m * T_SZ * I_DIMC);
    __syncthreads();

    v8f accR = {}, accZ = {}, accNI = {}, accNH = {};   // first WMMA uses inline C=0
    #pragma unroll
    for (int c = 0; c < 4; ++c) {                               // gi = x_t @ Wih^T
      v16h a  = *(const v16h*)(fcx + (c * 32 + lane) * 16);
      v16h br = *(const v16h*)(wb0 + (((wv    ) * 4 + c) * 32 + lane) * 16);
      v16h bz = *(const v16h*)(wb0 + (((wv + 4) * 4 + c) * 32 + lane) * 16);
      v16h bn = *(const v16h*)(wb0 + (((wv + 8) * 4 + c) * 32 + lane) * 16);
      accR  = WMMA(a, br, accR);
      accZ  = WMMA(a, bz, accZ);
      accNI = WMMA(a, bn, accNI);
    }
    #pragma unroll
    for (int c = 0; c < 2; ++c) {                               // gh = h @ Whh^T
      v16h a  = *(const v16h*)(hfr + (c * 32 + lane) * 16);
      v16h br = *(const v16h*)(wb1 + (((wv    ) * 2 + c) * 32 + lane) * 16);
      v16h bz = *(const v16h*)(wb1 + (((wv + 4) * 2 + c) * 32 + lane) * 16);
      v16h bn = *(const v16h*)(wb1 + (((wv + 8) * 2 + c) * 32 + lane) * 16);
      accR  = WMMA(a, br, accR);
      accZ  = WMMA(a, bz, accZ);
      accNH = WMMA(a, bn, accNH);
    }
    #pragma unroll
    for (int i = 0; i < 8; ++i) {                               // fused GRU gating (fp32)
      float r = fast_sigmoid(accR[i] + eb_r);
      float z = fast_sigmoid(accZ[i] + eb_z);
      float n = fast_tanh(accNI[i] + eb_ni + r * (accNH[i] + eb_nh));
      hprev[i] = n + z * (hprev[i] - n);
    }
    __syncthreads();                                            // all hfr reads done
    #pragma unroll
    for (int i = 0; i < 8; ++i) hdst[i * 16] = (_Float16)hprev[i];
  }
  __syncthreads();

  // feature_vec = final encoder h (output region 0: B x 64), straight from registers
  #pragma unroll
  for (int i = 0; i < 8; ++i)
    out[(size_t)(b0 + mhi + i) * H_DIMC + hcol] = hprev[i];

  // ---- stage decoder weights (reuse LDS regions) ----
  for (int idx = tid; idx < G3 * H_DIMC; idx += 128) {          // Wcomb[192][64]
    int n = idx >> 6, k = idx & 63;
    int l = (((k >> 4) & 1) << 4) | (n & 15);
    wb0[((((n >> 4) << 1) | (k >> 5)) * 32 + l) * 16 + (k & 15)] = (_Float16)wcomb[idx];
  }
  for (int idx = tid; idx < G3 * H_DIMC; idx += 128) {          // dec_Whh[192][64]
    int n = idx >> 6, k = idx & 63;
    int l = (((k >> 4) & 1) << 4) | (n & 15);
    wb1[((((n >> 4) << 1) | (k >> 5)) * 32 + l) * 16 + (k & 15)] = (_Float16)dWhh[idx];
  }
  for (int idx = tid; idx < I_DIMC * H_DIMC; idx += 128) {      // fc_W[128][64]: N=128(8 tiles), K=64
    int j = idx >> 6, k = idx & 63;
    int l = (((k >> 4) & 1) << 4) | (j & 15);
    fcx[((((j >> 4) << 1) | (k >> 5)) * 32 + l) * 16 + (k & 15)] = (_Float16)fcW[idx];
  }
  for (int idx = tid; idx < G3; idx += 128) { bI[idx] = bcomb[idx]; bH[idx] = dbhh[idx]; bI0[idx] = dbih[idx]; }
  for (int idx = tid; idx < I_DIMC; idx += 128) fb[idx] = fcb[idx];
  __syncthreads();

  // decoder biases hoisted into registers: t==0 (input zeros -> dec_bih) vs t>0 (bcomb)
  const float d0_r  = bI0[hcol]       + bH[hcol];
  const float d0_z  = bI0[64 + hcol]  + bH[64 + hcol];
  const float d0_ni = bI0[128 + hcol];
  const float d1_r  = bI[hcol]        + bH[hcol];
  const float d1_z  = bI[64 + hcol]   + bH[64 + hcol];
  const float d1_ni = bI[128 + hcol];
  const float d_nh  = bH[128 + hcol];
  float fbv[2];
  fbv[0] = fb[((wv << 1) << 4) | nsub];
  fbv[1] = fb[(((wv << 1) | 1) << 4) | nsub];

  // ================= decoder: 512 sequential autoregressive steps =================
  const size_t rec = (size_t)B_SZ * H_DIMC;   // reconstructed offset in d_out
  for (int t = 0; t < T_SZ; ++t) {
    const float b_r  = (t == 0) ? d0_r  : d1_r;
    const float b_z  = (t == 0) ? d0_z  : d1_z;
    const float b_ni = (t == 0) ? d0_ni : d1_ni;

    v8f accR = {}, accZ = {}, accNI = {}, accNH = {};
    #pragma unroll
    for (int c = 0; c < 2; ++c) {
      v16h a  = *(const v16h*)(hfr + (c * 32 + lane) * 16);
      v16h hr = *(const v16h*)(wb1 + (((wv    ) * 2 + c) * 32 + lane) * 16);
      v16h hz = *(const v16h*)(wb1 + (((wv + 4) * 2 + c) * 32 + lane) * 16);
      v16h hn = *(const v16h*)(wb1 + (((wv + 8) * 2 + c) * 32 + lane) * 16);
      accR  = WMMA(a, hr, accR);
      accZ  = WMMA(a, hz, accZ);
      accNH = WMMA(a, hn, accNH);
      if (t > 0) {                            // gi = h_{t-1} @ Wcomb^T (+bcomb in bias)
        v16h ir = *(const v16h*)(wb0 + (((wv    ) * 2 + c) * 32 + lane) * 16);
        v16h iz = *(const v16h*)(wb0 + (((wv + 4) * 2 + c) * 32 + lane) * 16);
        v16h in_= *(const v16h*)(wb0 + (((wv + 8) * 2 + c) * 32 + lane) * 16);
        accR  = WMMA(a, ir, accR);
        accZ  = WMMA(a, iz, accZ);
        accNI = WMMA(a, in_, accNI);
      }
    }
    #pragma unroll
    for (int i = 0; i < 8; ++i) {
      float r = fast_sigmoid(accR[i] + b_r);
      float z = fast_sigmoid(accZ[i] + b_z);
      float n = fast_tanh(accNI[i] + b_ni + r * (accNH[i] + d_nh));
      hprev[i] = n + z * (hprev[i] - n);
    }
    __syncthreads();                          // all hfr reads (gates + prev FC) done
    #pragma unroll
    for (int i = 0; i < 8; ++i) hdst[i * 16] = (_Float16)hprev[i];
    __syncthreads();                          // h_t visible to all waves

    // out_t = h_t @ fc_W^T + fc_b  -> reconstructed[b, t, :]  (wave handles 2 of 8 N-tiles)
    #pragma unroll
    for (int jt = 0; jt < 2; ++jt) {
      int tile = (wv << 1) | jt;
      int jcol = (tile << 4) | nsub;
      v8f o = {};                             // inline C=0; fc_b added at store
      #pragma unroll
      for (int c = 0; c < 2; ++c) {
        v16h a  = *(const v16h*)(hfr + (c * 32 + lane) * 16);
        v16h bw = *(const v16h*)(fcx + ((tile * 2 + c) * 32 + lane) * 16);
        o = WMMA(a, bw, o);
      }
      #pragma unroll
      for (int i = 0; i < 8; ++i) {
        __builtin_nontemporal_store(
            o[i] + fbv[jt],
            out + rec + ((size_t)(b0 + mhi + i) * T_SZ + t) * I_DIMC + jcol);
      }
    }
  }
}

// tiny classifier: feat(512x64) -> relu(W1)+b1 -> W2+b2 -> logits(512x2)
__global__ void gruae_cls(const float* __restrict__ feat,
                          const float* __restrict__ W1, const float* __restrict__ b1,
                          const float* __restrict__ W2, const float* __restrict__ b2,
                          float* __restrict__ logits) {
  int b = blockIdx.x * blockDim.x + threadIdx.x;
  if (b >= B_SZ) return;
  float hd[32];
  for (int j = 0; j < 32; ++j) {
    float a = b1[j];
    for (int k = 0; k < H_DIMC; ++k) a += feat[b * H_DIMC + k] * W1[j * H_DIMC + k];
    hd[j] = fmaxf(a, 0.0f);
  }
  for (int o = 0; o < 2; ++o) {
    float a = b2[o];
    for (int k = 0; k < 32; ++k) a += hd[k] * W2[o * 32 + k];
    logits[b * 2 + o] = a;
  }
}

extern "C" void kernel_launch(void* const* d_in, const int* in_sizes, int n_in,
                              void* d_out, int out_size, void* d_ws, size_t ws_size,
                              hipStream_t stream) {
  const float* x    = (const float*)d_in[0];
  const float* eWih = (const float*)d_in[1];
  const float* eWhh = (const float*)d_in[2];
  const float* ebih = (const float*)d_in[3];
  const float* ebhh = (const float*)d_in[4];
  const float* dWih = (const float*)d_in[5];
  const float* dWhh = (const float*)d_in[6];
  const float* dbih = (const float*)d_in[7];
  const float* dbhh = (const float*)d_in[8];
  const float* fcW  = (const float*)d_in[9];
  const float* fcb  = (const float*)d_in[10];
  const float* cW1  = (const float*)d_in[11];
  const float* cb1  = (const float*)d_in[12];
  const float* cW2  = (const float*)d_in[13];
  const float* cb2  = (const float*)d_in[14];
  float* out = (float*)d_out;

  float* wcomb = (float*)d_ws;            // 192*64 f32
  float* bcomb = wcomb + G3 * H_DIMC;     // 192 f32

  gruae_prep<<<dim3(G3), dim3(H_DIMC), 0, stream>>>(dWih, fcW, fcb, dbih, wcomb, bcomb);
  gruae_main<<<dim3(B_SZ / BTILE), dim3(128), L_TOTAL, stream>>>(
      x, eWih, eWhh, ebih, ebhh, dWhh, dbih, dbhh, fcW, fcb, wcomb, bcomb, out);
  size_t logit_off = (size_t)B_SZ * H_DIMC + (size_t)B_SZ * T_SZ * I_DIMC;
  gruae_cls<<<dim3(4), dim3(128), 0, stream>>>(out, cW1, cb1, cW2, cb2, out + logit_off);
}
